// FSTBlock_53575422050390
// MI455X (gfx1250) — matrix operand order repaired and checked by reading, not compiled, not run on hardware
//
#include <hip/hip_runtime.h>
#include <cstddef>

// ---------------------------------------------------------------------------
// MI455X (gfx1250) factorized space-time transformer block.
// All GEMMs + attention matmuls run on v_wmma_f32_16x16x32_bf16 (wave32).
// ---------------------------------------------------------------------------

typedef unsigned short ushort_t;
typedef __attribute__((ext_vector_type(16))) __bf16 v16bf;
typedef __attribute__((ext_vector_type(8)))  float  v8f;

union BF16x16 { v16bf v; uint4 u[2]; };

#define C_DIM   768
#define QKV_LD  2304
#define NHEAD   12
#define M_ROWS  25088          // B*T*HW = 16*8*196
#define ATT_SCALE 0.125f       // 64^-0.5
#define LNEPS   1e-5f

__device__ __forceinline__ ushort_t f2bf(float f) {
  unsigned u = __float_as_uint(f);
  u += 0x7FFFu + ((u >> 16) & 1u);        // round-to-nearest-even
  return (ushort_t)(u >> 16);
}
__device__ __forceinline__ float gelu_exact(float x) {
  return 0.5f * x * (1.0f + erff(x * 0.70710678118654752f));
}
// temporal row (b, hw, t) -> original row (b, t, hw)
__device__ __forceinline__ long long remap_row(int r) {
  int b = r / 1568; int rem = r - b * 1568;    // 1568 = HW*T
  int hw = rem >> 3; int t = rem & 7;
  return (long long)(b * 8 + t) * 196 + hw;
}

// ---------------------------------------------------------------------------
// f32 -> bf16 weight conversion
// ---------------------------------------------------------------------------
__global__ void cvt_f32_bf16(const float* __restrict__ in,
                             ushort_t* __restrict__ out, int n) {
  int i = blockIdx.x * blockDim.x + threadIdx.x;
  if (i < n) out[i] = f2bf(in[i]);
}

// ---------------------------------------------------------------------------
// LayerNorm (C=768) -> bf16.  mode 0: direct rows.  mode 1: output row is in
// temporal order, gather the input row from the original (b,t,hw) layout.
// ---------------------------------------------------------------------------
__global__ __launch_bounds__(256)
void ln_bf16(const float* __restrict__ in, const float* __restrict__ g,
             const float* __restrict__ bsh, ushort_t* __restrict__ out, int mode) {
  __shared__ float r1[256];
  __shared__ float r2[256];
  const int row = blockIdx.x;
  long long irow = row;
  if (mode == 1) irow = remap_row(row);
  const float* x = in + (size_t)irow * C_DIM;
  const int tid = threadIdx.x;
  float x0 = x[tid], x1 = x[tid + 256], x2 = x[tid + 512];
  r1[tid] = x0 + x1 + x2;
  r2[tid] = x0 * x0 + x1 * x1 + x2 * x2;
  __syncthreads();
  for (int off = 128; off > 0; off >>= 1) {
    if (tid < off) { r1[tid] += r1[tid + off]; r2[tid] += r2[tid + off]; }
    __syncthreads();
  }
  const float mu  = r1[0] * (1.0f / 768.0f);
  const float var = r2[0] * (1.0f / 768.0f) - mu * mu;
  const float inv = rsqrtf(var + LNEPS);
  ushort_t* o = out + (size_t)row * C_DIM;
  o[tid]       = f2bf((x0 - mu) * inv * g[tid]       + bsh[tid]);
  o[tid + 256] = f2bf((x1 - mu) * inv * g[tid + 256] + bsh[tid + 256]);
  o[tid + 512] = f2bf((x2 - mu) * inv * g[tid + 512] + bsh[tid + 512]);
}

// ---------------------------------------------------------------------------
// WMMA GEMM:  Y[M,N] = X[M,K] * W[N,K]^T (+ bias) (+ epilogue)
//   X bf16 row-major, W bf16 row-major (torch Linear layout -> B columns are
//   W rows, contiguous in K).
//   Block = 256 threads = 8 waves; block tile 128x128, wave grid 4(M) x 2(N);
//   each wave owns 32x64 -> 8 accumulators, 8 WMMAs per 32-wide K-step from
//   2 A-fragments x 4 B-fragments.  Double-buffered LDS: one barrier/K-step.
//   epi: 0 = bias, 1 = bias+exact GELU, 2 = bias+residual.
//   remap: store (and residual fetch) through temporal->orig row remap.
// ---------------------------------------------------------------------------
#define TLDS (128 * 40)        // one LDS tile: 128 rows x (32 K + 8 pad)

__global__ __launch_bounds__(256)
void gemm_wmma_bf16(const ushort_t* __restrict__ A, const ushort_t* __restrict__ W,
                    const float* __restrict__ bias, const float* __restrict__ res,
                    float* __restrict__ outF, ushort_t* __restrict__ outB,
                    int M, int N, int K, int epi, int remap, int outbf) {
  __shared__ __align__(16) ushort_t Al[2 * TLDS];
  __shared__ __align__(16) ushort_t Bl[2 * TLDS];
  const int tileM = blockIdx.y * 128;
  const int tileN = blockIdx.x * 128;
  const int tid  = threadIdx.x;
  const int wave = tid >> 5, lane = tid & 31;
  const int wm = wave >> 1, wn = wave & 1;
  const int ln = lane & 15, hi16 = lane >> 4;
  const int ksA = hi16 * 8;          // A fragment K pattern {ks..ks+7, ks+16..ks+23}
  const int kbB = hi16 * 8;          // B fragment K pattern {kb..kb+7, kb+16..kb+23}

  // cooperative loader: 128x32 tile = 512 uint4; 2 per thread (rows r, r+64)
  const int lr = tid >> 2, lc = (tid & 3) * 8;
  const ushort_t* aptr0 = A + (size_t)(tileM + lr) * K + lc;
  const ushort_t* aptr1 = aptr0 + (size_t)64 * K;
  const ushort_t* wptr0 = W + (size_t)(tileN + lr) * K + lc;
  const ushort_t* wptr1 = wptr0 + (size_t)64 * K;
  const int so0 = lr * 40 + lc, so1 = (lr + 64) * 40 + lc;

  v8f zero8 = {0.f, 0.f, 0.f, 0.f, 0.f, 0.f, 0.f, 0.f};
  v8f acc[2][4];
#pragma unroll
  for (int f = 0; f < 2; ++f)
#pragma unroll
    for (int s = 0; s < 4; ++s) acc[f][s] = zero8;

  int buf = 0;
  for (int kk = 0; kk < K; kk += 32, buf ^= 1) {
    uint4 a0 = *(const uint4*)(aptr0 + kk);
    uint4 a1 = *(const uint4*)(aptr1 + kk);
    uint4 w0 = *(const uint4*)(wptr0 + kk);
    uint4 w1 = *(const uint4*)(wptr1 + kk);
    if (kk + 32 < K) {                       // global_prefetch_b8 next K-tile
      __builtin_prefetch(aptr0 + kk + 32, 0, 1);
      __builtin_prefetch(aptr1 + kk + 32, 0, 1);
      __builtin_prefetch(wptr0 + kk + 32, 0, 1);
      __builtin_prefetch(wptr1 + kk + 32, 0, 1);
    }
    ushort_t* as = &Al[buf * TLDS];
    ushort_t* bs = &Bl[buf * TLDS];
    *(uint4*)&as[so0] = a0;
    *(uint4*)&as[so1] = a1;
    *(uint4*)&bs[so0] = w0;
    *(uint4*)&bs[so1] = w1;
    __syncthreads();   // double buffer: WAR hazard is 2 iterations away

    BF16x16 af[2];
#pragma unroll
    for (int f = 0; f < 2; ++f) {
      const int ar = (wm * 32 + f * 16 + ln) * 40;
      af[f].u[0] = *(const uint4*)&as[ar + ksA];
      af[f].u[1] = *(const uint4*)&as[ar + 16 + ksA];
    }
#pragma unroll
    for (int sub = 0; sub < 4; ++sub) {
      BF16x16 b;
      const int br = (wn * 64 + sub * 16 + ln) * 40;
      b.u[0] = *(const uint4*)&bs[br + kbB];
      b.u[1] = *(const uint4*)&bs[br + 16 + kbB];
      acc[0][sub] = __builtin_amdgcn_wmma_f32_16x16x32_bf16(false, af[0].v, false, b.v,
                                                            (short)0, acc[0][sub], false, false);
      acc[1][sub] = __builtin_amdgcn_wmma_f32_16x16x32_bf16(false, af[1].v, false, b.v,
                                                            (short)0, acc[1][sub], false, false);
    }
  }

  // Epilogue. C layout: lane holds col = ln, rows = i + 8*hi16.
#pragma unroll
  for (int f = 0; f < 2; ++f) {
    for (int sub = 0; sub < 4; ++sub) {
      const int col = tileN + wn * 64 + sub * 16 + ln;
      const float bv = bias ? bias[col] : 0.0f;
#pragma unroll
      for (int i = 0; i < 8; ++i) {
        const int row = tileM + wm * 32 + f * 16 + hi16 * 8 + i;
        float v = acc[f][sub][i] + bv;
        if (epi == 1) v = gelu_exact(v);
        long long orow = remap ? remap_row(row) : (long long)row;
        size_t off = (size_t)orow * N + col;
        if (epi == 2) v += res[off];
        if (outbf) outB[off] = f2bf(v);
        else       outF[off] = v;
      }
    }
  }
}

// ---------------------------------------------------------------------------
// Flash attention, WMMA bf16.  One block = 128 threads = 4 waves; each wave
// owns a 16-query tile; key/value blocks of 32 staged in LDS (V transposed).
// qkv layout per token row (stride 2304): [q(768) | k(768) | v(768)],
// each as H=12 heads x 64.  Works for seqN=196 (spatial) and 8 (temporal).
// ---------------------------------------------------------------------------
__global__ __launch_bounds__(128)
void attn_wmma(const ushort_t* __restrict__ qkv, ushort_t* __restrict__ out,
               int seqN) {
  __shared__ __align__(16) ushort_t Kl[32 * 72];       // 32 keys x 64 hd (+pad)
  __shared__ __align__(16) ushort_t Vt[64 * 40];       // 64 hd x 32 keys (+pad)
  __shared__ __align__(16) ushort_t Pl[4 * 16 * 40];   // per-wave P tile 16x32

  const int p = blockIdx.x;
  const int sIdx = p / NHEAD;
  const int h = p - sIdx * NHEAD;
  const size_t rowBase = (size_t)sIdx * seqN;
  const int qcol = h * 64, kcol = C_DIM + h * 64, vcol = 2 * C_DIM + h * 64;

  const int tid = threadIdx.x, wave = tid >> 5, lane = tid & 31;
  const int ln = lane & 15, hi16 = lane >> 4;
  const int ksA = hi16 * 8, kbB = hi16 * 8;
  const int q0 = (blockIdx.y * 4 + wave) * 16;
  const bool active = (q0 < seqN);

  // Q fragments (A-matrix, 2 K-slices of 32 over HD=64), clamped rows.
  BF16x16 aq[2];
  {
    int qrow = q0 + ln; if (qrow >= seqN) qrow = seqN - 1;
    const ushort_t* qp = qkv + (rowBase + qrow) * QKV_LD + qcol;
#pragma unroll
    for (int hs = 0; hs < 2; ++hs) {
      aq[hs].u[0] = *(const uint4*)(qp + hs * 32 + ksA);
      aq[hs].u[1] = *(const uint4*)(qp + hs * 32 + 16 + ksA);
    }
  }

  v8f zero8 = {0.f, 0.f, 0.f, 0.f, 0.f, 0.f, 0.f, 0.f};
  float mrow[8], lrow[8];
  v8f o[4]; o[0] = zero8; o[1] = zero8; o[2] = zero8; o[3] = zero8;
#pragma unroll
  for (int i = 0; i < 8; ++i) { mrow[i] = -3.0e38f; lrow[i] = 0.0f; }

  const int nkb = (seqN + 31) >> 5;
  for (int kb = 0; kb < nkb; ++kb) {
    // ---- stage 32 keys of K (row-major) and V (transposed) ----
#pragma unroll
    for (int part = 0; part < 2; ++part) {
      int idx = tid + part * 128;
      int key = idx >> 3, gg = idx & 7;
      int krow = kb * 32 + key; if (krow >= seqN) krow = seqN - 1;
      const ushort_t* kp = qkv + (rowBase + krow) * QKV_LD;
      uint4 kv = *(const uint4*)(kp + kcol + gg * 8);
      uint4 vv = *(const uint4*)(kp + vcol + gg * 8);
      *(uint4*)&Kl[key * 72 + gg * 8] = kv;
      const ushort_t* vs = (const ushort_t*)&vv;
#pragma unroll
      for (int j = 0; j < 8; ++j) Vt[(gg * 8 + j) * 40 + key] = vs[j];
    }
    __syncthreads();

    if (active) {
      // ---- S = Q * K^T  (16 queries x 32 keys, two 16-key subtiles) ----
      v8f sacc[2]; sacc[0] = zero8; sacc[1] = zero8;
#pragma unroll
      for (int hs = 0; hs < 2; ++hs) {
        BF16x16 bk0, bk1;
        bk0.u[0] = *(const uint4*)&Kl[ln * 72 + hs * 32 + kbB];
        bk0.u[1] = *(const uint4*)&Kl[ln * 72 + hs * 32 + 16 + kbB];
        bk1.u[0] = *(const uint4*)&Kl[(16 + ln) * 72 + hs * 32 + kbB];
        bk1.u[1] = *(const uint4*)&Kl[(16 + ln) * 72 + hs * 32 + 16 + kbB];
        sacc[0] = __builtin_amdgcn_wmma_f32_16x16x32_bf16(false, aq[hs].v, false, bk0.v,
                                                          (short)0, sacc[0], false, false);
        sacc[1] = __builtin_amdgcn_wmma_f32_16x16x32_bf16(false, aq[hs].v, false, bk1.v,
                                                          (short)0, sacc[1], false, false);
      }

      // ---- online softmax update (row reduce across 16-lane groups) ----
      float pa[8], pb[8], alpha[8];
      const int c0 = kb * 32 + ln, c1 = c0 + 16;
#pragma unroll
      for (int i = 0; i < 8; ++i) {
        float s0 = (c0 < seqN) ? sacc[0][i] * ATT_SCALE : -3.0e38f;
        float s1 = (c1 < seqN) ? sacc[1][i] * ATT_SCALE : -3.0e38f;
        float rm = fmaxf(s0, s1);
        rm = fmaxf(rm, __shfl_xor(rm, 1));
        rm = fmaxf(rm, __shfl_xor(rm, 2));
        rm = fmaxf(rm, __shfl_xor(rm, 4));
        rm = fmaxf(rm, __shfl_xor(rm, 8));
        float mnew = fmaxf(mrow[i], rm);
        alpha[i] = __expf(mrow[i] - mnew);
        mrow[i] = mnew;
        pa[i] = __expf(s0 - mnew);
        pb[i] = __expf(s1 - mnew);
        float rs = pa[i] + pb[i];
        rs += __shfl_xor(rs, 1); rs += __shfl_xor(rs, 2);
        rs += __shfl_xor(rs, 4); rs += __shfl_xor(rs, 8);
        lrow[i] = lrow[i] * alpha[i] + rs;
      }
#pragma unroll
      for (int t = 0; t < 4; ++t)
#pragma unroll
        for (int i = 0; i < 8; ++i) o[t][i] *= alpha[i];

      // ---- transpose P (C layout -> A layout) through per-wave LDS ----
      ushort_t* pw = &Pl[wave * 16 * 40];
#pragma unroll
      for (int i = 0; i < 8; ++i) {
        int pr = hi16 * 8 + i;
        pw[pr * 40 + ln]      = f2bf(pa[i]);
        pw[pr * 40 + 16 + ln] = f2bf(pb[i]);
      }
      asm volatile("" ::: "memory");   // keep compiler from reordering ds ops
      BF16x16 ap;
      ap.u[0] = *(const uint4*)&pw[ln * 40 + ksA];
      ap.u[1] = *(const uint4*)&pw[ln * 40 + 16 + ksA];

      // ---- O += P * V  (four 16-wide HD subtiles) ----
#pragma unroll
      for (int t = 0; t < 4; ++t) {
        BF16x16 bv;
        bv.u[0] = *(const uint4*)&Vt[(t * 16 + ln) * 40 + kbB];
        bv.u[1] = *(const uint4*)&Vt[(t * 16 + ln) * 40 + 16 + kbB];
        o[t] = __builtin_amdgcn_wmma_f32_16x16x32_bf16(false, ap.v, false, bv.v,
                                                       (short)0, o[t], false, false);
      }
    }
    __syncthreads();
  }

  if (active) {
#pragma unroll
    for (int i = 0; i < 8; ++i) {
      int row = q0 + hi16 * 8 + i;
      if (row < seqN) {
        float inv = 1.0f / lrow[i];
        ushort_t* op = out + (rowBase + row) * C_DIM + h * 64;
#pragma unroll
        for (int t = 0; t < 4; ++t) op[t * 16 + ln] = f2bf(o[t][i] * inv);
      }
    }
  }
}

// ---------------------------------------------------------------------------
// Workspace layout (bytes).  hmid (154.1MB) aliases the xs+qkv regions which
// are both dead by the time the MLP runs.
// ---------------------------------------------------------------------------
#define OFF_XS   0ull                       // f32 [25088,768]   77,070,336 B
#define OFF_QKV  77070336ull                // bf16 [25088,2304] 115,605,504 B
#define OFF_BFA  192675840ull               // bf16 [25088,768]  38,535,168 B
#define OFF_ATT  231211008ull               // bf16 [25088,768]  38,535,168 B
#define OFF_W    269746176ull               // bf16 weights      18,874,368 B

extern "C" void kernel_launch(void* const* d_in, const int* in_sizes, int n_in,
                              void* d_out, int out_size, void* d_ws, size_t ws_size,
                              hipStream_t stream) {
  const float* x      = (const float*)d_in[0];
  const float* sp_g   = (const float*)d_in[1];
  const float* sp_b   = (const float*)d_in[2];
  const float* Wqkv_s = (const float*)d_in[3];
  const float* Wp_s   = (const float*)d_in[4];
  const float* bp_s   = (const float*)d_in[5];
  const float* tm_g   = (const float*)d_in[6];
  const float* tm_b   = (const float*)d_in[7];
  const float* Wqkv_t = (const float*)d_in[8];
  const float* Wp_t   = (const float*)d_in[9];
  const float* bp_t   = (const float*)d_in[10];
  const float* fn_g   = (const float*)d_in[11];
  const float* fn_b   = (const float*)d_in[12];
  const float* W1     = (const float*)d_in[13];
  const float* b1     = (const float*)d_in[14];
  const float* W2     = (const float*)d_in[15];
  const float* b2     = (const float*)d_in[16];
  float* out = (float*)d_out;

  char* ws = (char*)d_ws;
  float*    xs    = (float*)   (ws + OFF_XS);
  ushort_t* qkv   = (ushort_t*)(ws + OFF_QKV);
  ushort_t* bfA   = (ushort_t*)(ws + OFF_BFA);
  ushort_t* attnO = (ushort_t*)(ws + OFF_ATT);
  ushort_t* wbuf  = (ushort_t*)(ws + OFF_W);
  ushort_t* hmid  = (ushort_t*)(ws + 0);     // [25088,3072] bf16, aliases xs+qkv

  ushort_t* wq_s = wbuf;
  ushort_t* wp_s = wbuf + 1769472;
  ushort_t* wq_t = wbuf + 2359296;
  ushort_t* wp_t = wbuf + 4128768;
  ushort_t* w1b  = wbuf + 4718592;
  ushort_t* w2b  = wbuf + 7077888;

  auto cv = [&](const float* src, ushort_t* dst, int n) {
    cvt_f32_bf16<<<(n + 255) / 256, 256, 0, stream>>>(src, dst, n);
  };
  cv(Wqkv_s, wq_s, 2304 * 768);
  cv(Wp_s,   wp_s, 768 * 768);
  cv(Wqkv_t, wq_t, 2304 * 768);
  cv(Wp_t,   wp_t, 768 * 768);
  cv(W1,     w1b,  3072 * 768);
  cv(W2,     w2b,  768 * 3072);

  dim3 b256(256), b128(128);
  dim3 g_qkv(2304 / 128, M_ROWS / 128);   // 18 x 196
  dim3 g_prj(768 / 128,  M_ROWS / 128);   //  6 x 196
  dim3 g_fc1(3072 / 128, M_ROWS / 128);   // 24 x 196
  dim3 g_as(128 * NHEAD, 4);              // spatial: 1536 pairs, 4 q-groups of 64
  dim3 g_at(3136 * NHEAD, 1);             // temporal: 37632 pairs, 1 q-group

  // ---- spatial attention branch ----
  ln_bf16<<<M_ROWS, b256, 0, stream>>>(x, sp_g, sp_b, bfA, 0);
  gemm_wmma_bf16<<<g_qkv, b256, 0, stream>>>(bfA, wq_s, nullptr, nullptr,
      nullptr, qkv, M_ROWS, 2304, 768, 0, 0, 1);
  attn_wmma<<<g_as, b128, 0, stream>>>(qkv, attnO, 196);
  gemm_wmma_bf16<<<g_prj, b256, 0, stream>>>(attnO, wp_s, bp_s, x,
      xs, nullptr, M_ROWS, 768, 768, 2, 0, 0);

  // ---- temporal attention branch (rows in (b,hw,t) order) ----
  ln_bf16<<<M_ROWS, b256, 0, stream>>>(xs, tm_g, tm_b, bfA, 1);
  gemm_wmma_bf16<<<g_qkv, b256, 0, stream>>>(bfA, wq_t, nullptr, nullptr,
      nullptr, qkv, M_ROWS, 2304, 768, 0, 0, 1);
  attn_wmma<<<g_at, b128, 0, stream>>>(qkv, attnO, 8);
  // proj + residual, scatter back to original (b,t,hw) layout -> d_out (= xo)
  gemm_wmma_bf16<<<g_prj, b256, 0, stream>>>(attnO, wp_t, bp_t, xs,
      out, nullptr, M_ROWS, 768, 768, 2, 1, 0);

  // ---- MLP ----
  ln_bf16<<<M_ROWS, b256, 0, stream>>>(out, fn_g, fn_b, bfA, 0);
  gemm_wmma_bf16<<<g_fc1, b256, 0, stream>>>(bfA, w1b, b1, nullptr,
      nullptr, hmid, M_ROWS, 3072, 768, 1, 0, 1);
  gemm_wmma_bf16<<<g_prj, b256, 0, stream>>>(hmid, w2b, b2, out,
      out, nullptr, M_ROWS, 768, 3072, 2, 0, 0);
}
